// GNNModel_71399536328787
// MI455X (gfx1250) — matrix operand order repaired and checked by reading, not compiled, not run on hardware
//
#include <hip/hip_runtime.h>
#include <hip/hip_bf16.h>

typedef __attribute__((ext_vector_type(16))) __bf16 v16bf;
typedef __attribute__((ext_vector_type(8)))  float  v8f;

#define NNODES 100000
#define NEDGES 1600000

// ---------------- utility kernels ----------------

__global__ void init_fill_kernel(float* p, float val, int n) {
    int i = blockIdx.x * blockDim.x + threadIdx.x;
    if (i < n) p[i] = val;
}

__global__ void accum_deg_kernel(const int* __restrict__ dst, float* __restrict__ deg, int e) {
    int i = blockIdx.x * blockDim.x + threadIdx.x;
    if (i < e) atomicAdd(&deg[dst[i]], 1.0f);
}

__global__ void rsqrt_kernel(float* p, int n) {
    int i = blockIdx.x * blockDim.x + threadIdx.x;
    if (i < n) p[i] = rsqrtf(p[i]);   // deg >= 1 always (self loops)
}

// ---------------- WMMA GEMM: Y[N,FOUT] = X[N,64] @ W[64,FOUT] ----------------
// One wave per 16x16 output tile; K = 64 -> 2x v_wmma_f32_16x16x32_bf16.

template <int FOUT>
__global__ void gemm_bf16_wmma_kernel(const float* __restrict__ X,
                                      const float* __restrict__ W,
                                      float* __restrict__ Y,
                                      int mtiles) {
    constexpr int KDIM   = 64;
    constexpr int NTILES = FOUT / 16;
    const int lane = threadIdx.x & 31;
    const int wave = threadIdx.x >> 5;
    const int tile = blockIdx.x * (blockDim.x >> 5) + wave;
    if (tile >= mtiles * NTILES) return;          // wave-uniform exit (EXEC safe)

    const int tm   = tile / NTILES;
    const int tn   = tile % NTILES;
    const int half = lane >> 4;                   // 0: lanes 0-15, 1: lanes 16-31
    const int l15  = lane & 15;
    const int arow = tm * 16 + l15;               // A-matrix row for this lane
    const int col  = tn * 16 + l15;               // B/C/D column for this lane

    v8f c = {0.f, 0.f, 0.f, 0.f, 0.f, 0.f, 0.f, 0.f};

#pragma unroll
    for (int kk = 0; kk < KDIM; kk += 32) {
        v16bf a, b;
#pragma unroll
        for (int p = 0; p < 8; ++p) {
            // A 16x32 bf16 layout: VGPR p holds K = {2p, 2p+1} (+16 for p>=4), +8 if half
            const int ka = kk + ((p < 4) ? (2 * p) : (16 + 2 * (p - 4))) + half * 8;
            a[2 * p]     = (__bf16)X[arow * KDIM + ka];
            a[2 * p + 1] = (__bf16)X[arow * KDIM + ka + 1];
            // B 32x16 bf16 layout: same K order as A: VGPR p holds K = {2p, 2p+1} (+16 for p>=4), +8 if half
            const int kb = kk + ((p < 4) ? (2 * p) : (16 + 2 * (p - 4))) + half * 8;
            b[2 * p]     = (__bf16)W[kb * FOUT + col];
            b[2 * p + 1] = (__bf16)W[(kb + 1) * FOUT + col];
        }
        c = __builtin_amdgcn_wmma_f32_16x16x32_bf16(
                /*neg_a=*/false, a, /*neg_b=*/false, b,
                /*c_mod=*/(short)0, c, /*reuse_a=*/false, /*reuse_b=*/false);
    }

#pragma unroll
    for (int r = 0; r < 8; ++r) {
        const int m = tm * 16 + r + 8 * half;     // C/D: VGPR r -> M = r + 8*half
        Y[m * FOUT + col] = c[r];
    }
}

// ---------------- edge scatter: out[dst] += dinv[src]*dinv[dst] * xw[src] ----------------
// One thread per (edge, 4-feature group): float4 gather + 4x global_atomic_add_f32.

template <int F>
__global__ void scatter_kernel(const float4* __restrict__ xw,
                               const int* __restrict__ src,
                               const int* __restrict__ dst,
                               const float* __restrict__ dinv,
                               float* __restrict__ out) {
    constexpr int G = F / 4;
    const long long i = (long long)blockIdx.x * blockDim.x + threadIdx.x;
    if (i >= (long long)NEDGES * G) return;
    const int g = (int)(i % G);
    const int e = (int)(i / G);
    const int s = src[e];
    const int d = dst[e];
    const float nrm = dinv[s] * dinv[d];
    const float4 v = xw[(long long)s * G + g];
    float* o = out + (long long)d * F + 4 * g;
    atomicAdd(o + 0, nrm * v.x);
    atomicAdd(o + 1, nrm * v.y);
    atomicAdd(o + 2, nrm * v.z);
    atomicAdd(o + 3, nrm * v.w);
}

// ---------------- layer-1 epilogue: self-loop + bias + ReLU (in place on h) ----------------

__global__ void finalize_relu_kernel(const float* __restrict__ xw,
                                     const float* __restrict__ dinv,
                                     const float* __restrict__ bias,
                                     float* __restrict__ h,
                                     int total, int F) {
    int i = blockIdx.x * blockDim.x + threadIdx.x;
    if (i >= total) return;
    const int node = i / F;
    const int f    = i % F;
    const float di = dinv[node];
    const float v  = h[i] + di * di * xw[i] + bias[f];
    h[i] = fmaxf(v, 0.0f);
}

// ---------------- layer-2 epilogue + mean pool: 8 nodes x 32 feats per block ----------------

__global__ void finalize_pool_kernel(const float* __restrict__ xw,
                                     const float* __restrict__ hagg,
                                     const float* __restrict__ dinv,
                                     const float* __restrict__ bias,
                                     float* __restrict__ pooled) {
    __shared__ float s[256];
    const int t    = threadIdx.x;
    const int f    = t & 31;
    const int node = blockIdx.x * 8 + (t >> 5);
    const float di = dinv[node];
    const int idx  = node * 32 + f;
    s[t] = hagg[idx] + di * di * xw[idx] + bias[f];
    __syncthreads();
    if (t < 128) s[t] += s[t + 128];
    __syncthreads();
    if (t < 64)  s[t] += s[t + 64];
    __syncthreads();
    if (t < 32)  atomicAdd(&pooled[f], s[t] + s[t + 32]);
}

// ---------------- log_softmax over 32 pooled values ----------------

__global__ void logsoftmax_kernel(const float* __restrict__ pooled, float* __restrict__ out) {
    __shared__ float vals[32];
    __shared__ float red[2];
    const int t = threadIdx.x;           // 32 threads = 1 wave
    const float v = pooled[t] * (1.0f / (float)NNODES);
    vals[t] = v;
    __syncthreads();
    if (t == 0) {
        float m = vals[0];
        for (int i = 1; i < 32; ++i) m = fmaxf(m, vals[i]);
        float sum = 0.f;
        for (int i = 0; i < 32; ++i) sum += __expf(vals[i] - m);
        red[0] = m;
        red[1] = logf(sum);
    }
    __syncthreads();
    out[t] = v - red[0] - red[1];
}

// ---------------- host-side launch sequence ----------------

extern "C" void kernel_launch(void* const* d_in, const int* in_sizes, int n_in,
                              void* d_out, int out_size, void* d_ws, size_t ws_size,
                              hipStream_t stream) {
    const float* x   = (const float*)d_in[0];
    const int*   ei  = (const int*)d_in[1];
    const int*   src = ei;
    const int*   dst = ei + NEDGES;
    const float* W1  = (const float*)d_in[2];
    const float* b1  = (const float*)d_in[3];
    const float* W2  = (const float*)d_in[4];
    const float* b2  = (const float*)d_in[5];
    float* out = (float*)d_out;

    float* ws     = (float*)d_ws;
    float* dinv   = ws;                               // N floats (deg -> dinv in place)
    float* bufA   = ws + NNODES;                      // N*64 floats (xw1, then xw2 in first N*32)
    float* bufB   = bufA + (size_t)NNODES * 64;       // N*64 floats (h1 agg, then h2 agg)
    float* pooled = bufB + (size_t)NNODES * 64;       // 32 floats

    const int B = 256;
    const int mtiles = NNODES / 16;                   // 6250 exactly

    // degree + symmetric normalization
    init_fill_kernel<<<(NNODES + B - 1) / B, B, 0, stream>>>(dinv, 1.0f, NNODES);
    accum_deg_kernel<<<(NEDGES + B - 1) / B, B, 0, stream>>>(dst, dinv, NEDGES);
    rsqrt_kernel<<<(NNODES + B - 1) / B, B, 0, stream>>>(dinv, NNODES);

    // ---- layer 1 ----
    {   // xw1 = x @ W1 : 6250*4 tiles / 8 waves per block = 3125 blocks
        const int tiles = mtiles * (64 / 16);
        gemm_bf16_wmma_kernel<64><<<(tiles + 7) / 8, B, 0, stream>>>(x, W1, bufA, mtiles);
    }
    init_fill_kernel<<<((NNODES * 64) + B - 1) / B, B, 0, stream>>>(bufB, 0.0f, NNODES * 64);
    {
        const long long total = (long long)NEDGES * 16;   // E * F/4
        scatter_kernel<64><<<(unsigned)((total + B - 1) / B), B, 0, stream>>>(
            (const float4*)bufA, src, dst, dinv, bufB);
    }
    finalize_relu_kernel<<<((NNODES * 64) + B - 1) / B, B, 0, stream>>>(
        bufA, dinv, b1, bufB, NNODES * 64, 64);

    // ---- layer 2 ----
    {   // xw2 = h1 @ W2 : 6250*2 tiles -> 1563 blocks (tail waves exit uniformly)
        const int tiles = mtiles * (32 / 16);
        gemm_bf16_wmma_kernel<32><<<(tiles + 7) / 8, B, 0, stream>>>(bufB, W2, bufA, mtiles);
    }
    // reuse bufB (h1 dead after GEMM, stream-ordered) as h2 aggregation buffer
    init_fill_kernel<<<((NNODES * 32) + B - 1) / B, B, 0, stream>>>(bufB, 0.0f, NNODES * 32);
    init_fill_kernel<<<1, 32, 0, stream>>>(pooled, 0.0f, 32);
    {
        const long long total = (long long)NEDGES * 8;    // E * F/4
        scatter_kernel<32><<<(unsigned)((total + B - 1) / B), B, 0, stream>>>(
            (const float4*)bufA, src, dst, dinv, bufB);
    }
    finalize_pool_kernel<<<NNODES / 8, B, 0, stream>>>(bufA, bufB, dinv, b2, pooled);

    // mean + log_softmax -> d_out[32]
    logsoftmax_kernel<<<1, 32, 0, stream>>>(pooled, out);
}
